// NLBlock1_34815004901745
// MI455X (gfx1250) — matrix operand order repaired and checked by reading, not compiled, not run on hardware
//
#include <hip/hip_runtime.h>
#include <hip/hip_bf16.h>

// Problem constants (from reference)
#define BB   64
#define CC   512
#define CRR  128
#define HWSZ 1024
#define NPP  4
#define KPIX 256

typedef __attribute__((ext_vector_type(16))) _Float16 v16h;
typedef __attribute__((ext_vector_type(8)))  _Float16 v8h;
typedef __attribute__((ext_vector_type(8)))  float    v8f;

// ---- workspace layout (offsets in f16 elements) ----
// WP : packed w_phi/w_theta/w_g  f16 [3][128][512]          = 196608
// WM : packed w_mask             f16 [512][128]             =  65536
// P  : phi   f16 [b][p][pix][cr]                            = 8388608
// T  : theta f16 [b][p][pix][cr]                            = 8388608
// G  : g     f16 [b][p][cr][pix]                            = 8388608
// O  : out   f16 [b][hw][cr]                                = 8388608
// then float region: partial sums [b][p][16 ktiles] (4096) + gate [b][p] (256)
#define WS_WP 0u
#define WS_WM 196608u
#define WS_P  262144u
#define WS_T  8650752u
#define WS_G  17039360u
#define WS_O  25427968u
#define WS_HALFS 33816576u   // ~67.6 MB of f16 + 17 KB of f32

// ---------------- WMMA fragment helpers (ISA VGPR layouts) ----------------
// A-matrix 16x32 f16 (MxK), src is row-major [M][K] with row stride `stride`.
// Lane L: M = L&15. halfs h=0..7 -> K = (L>=16?8:0)+h ; h=8..15 -> K = 16+(L>=16?8:0)+(h-8)
__device__ __forceinline__ v16h frag_a(const _Float16* row0, int stride, int lane) {
  const _Float16* rp = row0 + (lane & 15) * stride + ((lane >> 4) << 3);
  v8h lo = *(const v8h*)(rp);
  v8h hi = *(const v8h*)(rp + 16);
  v16h r;
#pragma unroll
  for (int i = 0; i < 8; ++i) { r[i] = lo[i]; r[i + 8] = hi[i]; }
  return r;
}
// B-matrix 32x16 f16 (KxN), src is [N][K] row-major (K contiguous), row stride `stride`.
// Lane L: N = L&15. halfs h=0..7 -> K = (L>=16?8:0)+h ; h=8..15 -> K = 16+(L>=16?8:0)+(h-8)
__device__ __forceinline__ v16h frag_b(const _Float16* row0, int stride, int lane) {
  const _Float16* rp = row0 + (lane & 15) * stride + ((lane >> 4) << 3);
  v8h lo = *(const v8h*)(rp);
  v8h hi = *(const v8h*)(rp + 16);
  v16h r;
#pragma unroll
  for (int i = 0; i < 8; ++i) { r[i] = lo[i]; r[i + 8] = hi[i]; }
  return r;
}
__device__ __forceinline__ v8f wmma_f16(v16h a, v16h b, v8f c) {
  return __builtin_amdgcn_wmma_f32_16x16x32_f16(false, a, false, b, (short)0, c, false, false);
}

// ---------------- Kernel 0: pack weights to f16 ----------------
__global__ void k_pack(const float* __restrict__ wphi, const float* __restrict__ wth,
                       const float* __restrict__ wg,   const float* __restrict__ wm,
                       _Float16* __restrict__ ws) {
  int gid = blockIdx.x * blockDim.x + threadIdx.x;
#pragma unroll
  for (int e = 0; e < 4; ++e) {
    int idx = gid * 4 + e;  // 0 .. 262143
    float v; unsigned dst;
    if (idx < 65536)        { v = wphi[idx];          dst = WS_WP + idx; }
    else if (idx < 131072)  { v = wth [idx - 65536];  dst = WS_WP + idx; }
    else if (idx < 196608)  { v = wg  [idx - 131072]; dst = WS_WP + idx; }
    else                    { v = wm  [idx - 196608]; dst = WS_WM + (idx - 196608); }
    ws[dst] = (_Float16)v;
  }
}

// ---------------- Kernel 1: projections (phi/theta/g), fused xc=xa+xb ----------------
// grid (8 hw-tiles, 3 which, 64 b) -> all blocks touching one batch's 4MB input slice
// are launch-adjacent, so xa/xb cross HBM once and hit L2 for the other reads.
// GEMM: M=CR(128), N=128 pixels, K=C(512). Block 256 (8 waves).
template <int WHICH>
__device__ __forceinline__ void proj_body(const float* __restrict__ xa,
                                          const float* __restrict__ xb,
                                          _Float16* __restrict__ ws, int nt0, int b) {
  __shared__ _Float16 Xs[128 * 40];  // [128 pix][32 chan + pad 8]
  int tid = threadIdx.x, lane = tid & 31, w = tid >> 5;
  const _Float16* Wp = ws + WS_WP + (size_t)WHICH * (CRR * CC);
  const float* xaB = xa + (size_t)b * CC * HWSZ;
  const float* xbB = xb + (size_t)b * CC * HWSZ;
  v8f acc[8] = {};
  for (int kc = 0; kc < CC; kc += 32) {
    __syncthreads();
#pragma unroll
    for (int i = 0; i < 16; ++i) {
      int idx = i * 256 + tid;
      int cl = idx >> 7, nl = idx & 127;
      size_t gi = (size_t)(kc + cl) * HWSZ + nt0 + nl;
      float v;
      if (WHICH == 0)      v = xaB[gi];
      else if (WHICH == 1) v = xbB[gi];
      else                 v = xaB[gi] + xbB[gi];
      Xs[nl * 40 + cl] = (_Float16)v;
    }
    if (kc + 32 < CC) {
      const float* pf = (WHICH == 1) ? xbB : xaB;
      __builtin_prefetch(pf + (size_t)(kc + 32 + (tid >> 3)) * HWSZ + nt0 + ((tid & 7) << 4), 0, 1);
    }
    __syncthreads();
    // preload A + all 8 B fragments -> one ds clause + single wait -> dense WMMA burst
    v16h a = frag_a(Wp + (size_t)(w * 16) * CC + kc, CC, lane);
    v16h bf[8];
#pragma unroll
    for (int nt = 0; nt < 8; ++nt) bf[nt] = frag_b(Xs + nt * 16 * 40, 40, lane);
#pragma unroll
    for (int nt = 0; nt < 8; ++nt) acc[nt] = wmma_f16(a, bf[nt], acc[nt]);
  }
  int crb = w * 16;
#pragma unroll
  for (int nt = 0; nt < 8; ++nt) {
#pragma unroll
    for (int r = 0; r < 8; ++r) {
      int cr = crb + ((lane < 16) ? r : r + 8);
      int hw = nt0 + nt * 16 + (lane & 15);
      int h = hw >> 5, wp = hw & 31;
      int p = ((h >> 4) << 1) | (wp >> 4);
      int k = ((h & 15) << 4) | (wp & 15);
      _Float16 v = (_Float16)acc[nt][r];
      if (WHICH == 0)      ws[WS_P + ((size_t)(b * 4 + p) * KPIX + k) * CRR + cr] = v;
      else if (WHICH == 1) ws[WS_T + ((size_t)(b * 4 + p) * KPIX + k) * CRR + cr] = v;
      else                 ws[WS_G + ((size_t)(b * 4 + p) * CRR + cr) * KPIX + k] = v;
    }
  }
}

__global__ void k_proj(const float* __restrict__ xa, const float* __restrict__ xb,
                       _Float16* __restrict__ ws) {
  int nt0 = blockIdx.x * 128;
  int which = blockIdx.y;
  int b = blockIdx.z;
  if (which == 0)      proj_body<0>(xa, xb, ws, nt0, b);
  else if (which == 1) proj_body<1>(xa, xb, ws, nt0, b);
  else                 proj_body<2>(xa, xb, ws, nt0, b);
}

// ---------------- Kernel 2: att strip (recompute) -> per-strip partial sums ----------------
// grid (16 k-tiles, 64 b), block 256. Each wave owns one patch p = w>>1, 8 l-tiles.
__global__ void k_att_mean(const _Float16* __restrict__ ws, float* __restrict__ part) {
  __shared__ float partw[8];
  int tid = threadIdx.x, lane = tid & 31, w = tid >> 5;
  int kb = blockIdx.x * 16;
  int b = blockIdx.y;
  int p = w >> 1;
  const _Float16* Tb = ws + WS_T + ((size_t)(b * 4 + p) * KPIX + kb) * CRR;
  const _Float16* Pb = ws + WS_P + (size_t)(b * 4 + p) * KPIX * CRR;
  v8f acc[8] = {};
  for (int crk = 0; crk < CRR; crk += 32) {
    v16h a = frag_a(Tb + crk, CRR, lane);
    v16h bf[8];
#pragma unroll
    for (int t = 0; t < 8; ++t) {
      int lt = (w & 1) * 8 + t;
      bf[t] = frag_b(Pb + (size_t)lt * 16 * CRR + crk, CRR, lane);
    }
#pragma unroll
    for (int t = 0; t < 8; ++t) acc[t] = wmma_f16(a, bf[t], acc[t]);
  }
  float local = 0.f;
#pragma unroll
  for (int t = 0; t < 8; ++t)
#pragma unroll
    for (int r = 0; r < 8; ++r) local += acc[t][r];
  for (int o = 16; o > 0; o >>= 1) local += __shfl_down(local, o, 32);
  if (lane == 0) partw[w] = local;
  __syncthreads();
  if (tid < 4) part[(b * 4 + tid) * 16 + blockIdx.x] = partw[2 * tid] + partw[2 * tid + 1];
}

// ---------------- Kernel 3: SE MLP (4 -> 2 -> 4), write gate + y tail of d_out ----------------
__global__ void k_se(const float* __restrict__ part, const float* __restrict__ w1,
                     const float* __restrict__ w2, float* __restrict__ gate,
                     float* __restrict__ out_tail) {
  __shared__ float y0s[256];
  int t = threadIdx.x;  // == b*4+p
  float s = 0.f;
#pragma unroll
  for (int i = 0; i < 16; ++i) s += part[t * 16 + i];
  y0s[t] = s * (1.0f / 65536.0f);
  __syncthreads();
  int b = t >> 2, p = t & 3;
  float h0 = 0.f, h1 = 0.f;
#pragma unroll
  for (int q = 0; q < 4; ++q) {
    float yq = y0s[b * 4 + q];
    h0 += yq * w1[q];
    h1 += yq * w1[4 + q];
  }
  h0 = fmaxf(h0, 0.f); h1 = fmaxf(h1, 0.f);
  float val = h0 * w2[p * 2 + 0] + h1 * w2[p * 2 + 1];
  float y = 1.f / (1.f + __expf(-val));
  gate[t] = y;
  out_tail[t] = y;
}

// ---------------- Kernel 4: gated att strip -> softmax over patches -> out GEMM ----------------
// grid (16 k-tiles, 64 b), block 256.
__global__ void k_softmax_out(_Float16* __restrict__ ws, const float* __restrict__ gate) {
  __shared__ float    attf[4 * 16 * 256];  // 64 KB
  __shared__ _Float16 atth[4 * 16 * 256];  // 32 KB
  int tid = threadIdx.x, lane = tid & 31, w = tid >> 5;
  int kb = blockIdx.x * 16;
  int b = blockIdx.y;
  int p = w >> 1;
  // GEMM1: att[k,l] = sum_cr theta^T[k,cr] * phi[cr,l]
  {
    const _Float16* Tb = ws + WS_T + ((size_t)(b * 4 + p) * KPIX + kb) * CRR;
    const _Float16* Pb = ws + WS_P + (size_t)(b * 4 + p) * KPIX * CRR;
    v8f acc[8] = {};
    for (int crk = 0; crk < CRR; crk += 32) {
      v16h a = frag_a(Tb + crk, CRR, lane);
      v16h bf[8];
#pragma unroll
      for (int t = 0; t < 8; ++t) {
        int lt = (w & 1) * 8 + t;
        bf[t] = frag_b(Pb + (size_t)lt * 16 * CRR + crk, CRR, lane);
      }
#pragma unroll
      for (int t = 0; t < 8; ++t) acc[t] = wmma_f16(a, bf[t], acc[t]);
    }
    float gt = gate[b * 4 + p];
#pragma unroll
    for (int t = 0; t < 8; ++t) {
      int l0 = ((w & 1) * 8 + t) * 16 + (lane & 15);
#pragma unroll
      for (int r = 0; r < 8; ++r) {
        int kl = (lane < 16) ? r : r + 8;
        attf[p * 4096 + kl * 256 + l0] = acc[t][r] * gt;
      }
    }
  }
  __syncthreads();
  // softmax across the 4 patches (dim=1 of [b,NP,K,K]) for each (k,l)
#pragma unroll
  for (int i = 0; i < 16; ++i) {
    int pos = i * 256 + tid;
    int kl = pos >> 8, l = pos & 255;
    float a0 = attf[0 * 4096 + kl * 256 + l];
    float a1 = attf[1 * 4096 + kl * 256 + l];
    float a2 = attf[2 * 4096 + kl * 256 + l];
    float a3 = attf[3 * 4096 + kl * 256 + l];
    float m = fmaxf(fmaxf(a0, a1), fmaxf(a2, a3));
    float e0 = __expf(a0 - m), e1 = __expf(a1 - m), e2 = __expf(a2 - m), e3 = __expf(a3 - m);
    float inv = 1.f / (e0 + e1 + e2 + e3);
    atth[0 * 4096 + kl * 256 + l] = (_Float16)(e0 * inv);
    atth[1 * 4096 + kl * 256 + l] = (_Float16)(e1 * inv);
    atth[2 * 4096 + kl * 256 + l] = (_Float16)(e2 * inv);
    atth[3 * 4096 + kl * 256 + l] = (_Float16)(e3 * inv);
  }
  __syncthreads();
  // GEMM2: out[k,cr] = sum_l att[k,l] * g[l,cr]  (G stored as [cr][l] -> B layout direct)
  v8f acc2[4] = {};
  for (int lb = 0; lb < KPIX; lb += 32) {
    v16h a = frag_a(atth + p * 4096 + lb, 256, lane);
    v16h bf[4];
#pragma unroll
    for (int t = 0; t < 4; ++t) {
      int crt = (w & 1) * 4 + t;
      bf[t] = frag_b(ws + WS_G + ((size_t)(b * 4 + p) * CRR + crt * 16) * KPIX + lb, KPIX, lane);
    }
#pragma unroll
    for (int t = 0; t < 4; ++t) acc2[t] = wmma_f16(a, bf[t], acc2[t]);
  }
  // fold during store: O[b][hw][cr] f16
#pragma unroll
  for (int t = 0; t < 4; ++t) {
    int cr = ((w & 1) * 4 + t) * 16 + (lane & 15);
#pragma unroll
    for (int r = 0; r < 8; ++r) {
      int kl = (lane < 16) ? r : r + 8;
      int kg = kb + kl;
      int h = ((p >> 1) << 4) + (kg >> 4);
      int wp = ((p & 1) << 4) + (kg & 15);
      int hw = h * 32 + wp;
      ws[WS_O + ((size_t)b * HWSZ + hw) * CRR + cr] = (_Float16)acc2[t][r];
    }
  }
}

// ---------------- Kernel 5: mask conv (C x CR GEMM) + residual xa+xb ----------------
// grid (8 hw-tiles, 4 c-blocks, 64 b), block 256.
// The B tile (O[hw][cr] slice) is shared by all 8 waves, so stage it once per block
// into LDS using the CDNA5 async global->LDS path (ASYNCcnt, no VGPR round-trip),
// then feed WMMA B fragments from LDS.
__global__ void k_mask(const _Float16* __restrict__ ws, const float* __restrict__ xa,
                       const float* __restrict__ xb, float* __restrict__ out) {
  __shared__ _Float16 Bs[128 * 40];  // [128 hw][32 cr + pad 8], row = 80B (16B-chunk aligned)
  int tid = threadIdx.x, lane = tid & 31, w = tid >> 5;
  int nt0 = blockIdx.x * 128;
  int cb = blockIdx.y * 128;
  int b = blockIdx.z;
  const _Float16* Wm = ws + WS_WM;
  const _Float16* Ob = ws + WS_O + (size_t)b * HWSZ * CRR;
  unsigned lds0 = (unsigned)(size_t)(&Bs[0]);  // LDS byte offset (generic addr low bits)
  v8f acc[8] = {};
  for (int crk = 0; crk < CRR; crk += 32) {
    __syncthreads();  // WAR: previous iteration's ds reads done before overwrite
#pragma unroll
    for (int j = 0; j < 2; ++j) {
      int idx = j * 256 + tid;          // 512 x 16B chunks = 8 KB tile
      int row = idx >> 2, chunk = idx & 3;
      unsigned loff = lds0 + row * 80 + chunk * 16;
      const char* gp = (const char*)(Ob + (size_t)(nt0 + row) * CRR + crk) + chunk * 16;
      asm volatile("global_load_async_to_lds_b128 %0, %1, off"
                   :: "v"(loff), "v"(gp) : "memory");
    }
    asm volatile("s_wait_asynccnt 0x0" ::: "memory");
    __syncthreads();
    v16h a = frag_a(Wm + (size_t)(cb + w * 16) * CRR + crk, CRR, lane);
    v16h bf[8];
#pragma unroll
    for (int nt = 0; nt < 8; ++nt) bf[nt] = frag_b(Bs + nt * 16 * 40, 40, lane);
#pragma unroll
    for (int nt = 0; nt < 8; ++nt) acc[nt] = wmma_f16(a, bf[nt], acc[nt]);
  }
#pragma unroll
  for (int nt = 0; nt < 8; ++nt) {
#pragma unroll
    for (int r = 0; r < 8; ++r) {
      int c = cb + w * 16 + ((lane < 16) ? r : r + 8);
      int hw = nt0 + nt * 16 + (lane & 15);
      size_t idx = (size_t)b * CC * HWSZ + (size_t)c * HWSZ + hw;
      out[idx] = acc[nt][r] + xa[idx] + xb[idx];
    }
  }
}

extern "C" void kernel_launch(void* const* d_in, const int* in_sizes, int n_in,
                              void* d_out, int out_size, void* d_ws, size_t ws_size,
                              hipStream_t stream) {
  const float* xa  = (const float*)d_in[0];
  const float* xb  = (const float*)d_in[1];
  const float* wph = (const float*)d_in[2];
  const float* wth = (const float*)d_in[3];
  const float* wg  = (const float*)d_in[4];
  const float* wm  = (const float*)d_in[5];
  const float* sw1 = (const float*)d_in[6];
  const float* sw2 = (const float*)d_in[7];
  _Float16* ws = (_Float16*)d_ws;
  float* fws  = (float*)((char*)d_ws + (size_t)WS_HALFS * 2);  // ~67.7 MB used
  float* part = fws;          // 4096 floats
  float* gate = fws + 4096;   // 256 floats
  float* out  = (float*)d_out;
  float* out_tail = out + (size_t)BB * CC * HWSZ;  // y output (256 floats)

  k_pack<<<256, 256, 0, stream>>>(wph, wth, wg, wm, ws);
  k_proj<<<dim3(8, 3, 64), 256, 0, stream>>>(xa, xb, ws);
  k_att_mean<<<dim3(16, 64), 256, 0, stream>>>(ws, part);
  k_se<<<1, 256, 0, stream>>>(part, sw1, sw2, gate, out_tail);
  k_softmax_out<<<dim3(16, 64), 256, 0, stream>>>(ws, gate);
  k_mask<<<dim3(8, 4, 64), 256, 0, stream>>>(ws, xa, xb, out);
}
